// ModulatedDeformConv2d_84774064489046
// MI455X (gfx1250) — matrix operand order repaired and checked by reading, not compiled, not run on hardware
//
#include <hip/hip_runtime.h>

typedef __attribute__((ext_vector_type(16))) __bf16 v16bf;
typedef __attribute__((ext_vector_type(8)))  float  v8f;

#define N_    8
#define C_    256
#define H_    64
#define W_    64
#define CO_   256
#define HO_   64
#define WO_   64
#define KK_   9
#define CK_   2304
#define HW_   4096

#define BM 64
#define BN 128
#define BK 32
#define NCHUNK (CK_/BK)   /* 72; reduction reordered ck' = k*C + c */

__device__ __forceinline__ unsigned short f2bf(float f) {
  union { float f; unsigned int u; } c; c.f = f;
  unsigned int u = c.u + 0x7FFFu + ((c.u >> 16) & 1u);   // round-to-nearest-even
  return (unsigned short)(u >> 16);
}

union BFU { unsigned short u; __bf16 b; };

// One-time repack: wpk[co][k*256 + c] = bf16(weight[co][c*9 + k])
__global__ __launch_bounds__(256)
void pack_weight_bf16(const float* __restrict__ w, unsigned short* __restrict__ wpk) {
  int i = blockIdx.x * 256 + threadIdx.x;      // over CO*CK (dst order)
  if (i >= CO_ * CK_) return;
  int co = i / CK_;
  int r  = i - co * CK_;
  int k  = r >> 8;                             // r / 256
  int c  = r & 255;
  wpk[i] = f2bf(w[co * CK_ + c * KK_ + k]);
}

__global__ __launch_bounds__(256)
void dcn_fused_wmma(const float* __restrict__ x,
                    const float* __restrict__ offset,
                    const float* __restrict__ mask,
                    const unsigned short* __restrict__ wpk,
                    float* __restrict__ out)
{
  __shared__ int            sIdx[BM*KK_][4];     // bilinear corner indices (clamped)
  __shared__ float          sWt [BM*KK_][4];     // corner weights * mask * valid
  __shared__ int            sXBase[BM];          // n*C*H*W
  __shared__ int            sOutBase[BM];        // n*CO*HO*WO + ho*WO + wo
  __shared__ unsigned short As[2][BM][BK + 8];   // im2col tile, bf16, double-buffered
  __shared__ unsigned short Bs[2][BN][BK + 8];   // weight tile (B^T rows), bf16

  const int tid     = threadIdx.x;
  const int pixBase = blockIdx.x * BM;
  const int coBase  = blockIdx.y * BN;

  // ---- per-pixel bases -------------------------------------------------
  if (tid < BM) {
    int p   = pixBase + tid;
    int n   = p >> 12;               // / (HO*WO)
    int rem = p & 4095;              // ho*WO + wo
    sXBase[tid]   = n << 20;         // n * C*H*W   (256*4096 = 1<<20)
    sOutBase[tid] = (n << 20) + rem; // n * CO*HO*WO + rem
  }

  // ---- bilinear sampling parameters for (pixel, k) ---------------------
  for (int i = tid; i < BM*KK_; i += 256) {
    int pl  = i / KK_;
    int k   = i - pl*KK_;
    int p   = pixBase + pl;
    int n   = p >> 12;
    int rem = p & 4095;
    int ho  = rem >> 6;
    int wo  = rem & 63;
    // offset layout [N][K][2][HO][WO], mask layout [N][K][HO][WO]
    float offy = offset[((n*18 + k*2) << 12) + rem];
    float offx = offset[((n*18 + k*2) << 12) + 4096 + rem];
    float mval = mask  [((n*9  + k  ) << 12) + rem];
    float py = (float)(k/3) + (float)ho - 1.0f + offy;
    float px = (float)(k%3) + (float)wo - 1.0f + offx;
    float fy = floorf(py), fx = floorf(px);
    float ly = py - fy,    lx = px - fx;
    int y0 = (int)fy, x0 = (int)fx;
    int y1 = y0 + 1,  x1 = x0 + 1;
    float v00 = (y0>=0 && y0<H_ && x0>=0 && x0<W_) ? 1.f : 0.f;
    float v01 = (y0>=0 && y0<H_ && x1>=0 && x1<W_) ? 1.f : 0.f;
    float v10 = (y1>=0 && y1<H_ && x0>=0 && x0<W_) ? 1.f : 0.f;
    float v11 = (y1>=0 && y1<H_ && x1>=0 && x1<W_) ? 1.f : 0.f;
    int yc0 = min(max(y0,0),H_-1), yc1 = min(max(y1,0),H_-1);
    int xc0 = min(max(x0,0),W_-1), xc1 = min(max(x1,0),W_-1);
    sIdx[i][0] = yc0*W_ + xc0;
    sIdx[i][1] = yc0*W_ + xc1;
    sIdx[i][2] = yc1*W_ + xc0;
    sIdx[i][3] = yc1*W_ + xc1;
    sWt[i][0] = (1.f-ly)*(1.f-lx)*mval*v00;
    sWt[i][1] = (1.f-ly)*lx      *mval*v01;
    sWt[i][2] = ly*(1.f-lx)      *mval*v10;
    sWt[i][3] = ly*lx            *mval*v11;
  }
  __syncthreads();

  const int lane = tid & 31;
  const int wv   = tid >> 5;    // wave 0..7: owns 16-wide N strip
  const int hs   = lane >> 4;   // half-wave select
  const int ln16 = lane & 15;

  const int arow  = tid >> 2;   // A-fill: pixel row (0..63)
  const int agrp  = tid & 3;    // A-fill: group of 8 channels
  const int brow  = tid >> 1;   // B-fill: co_local (0..127)
  const int bhalf = tid & 1;

  v8f acc[4];
  v8f zero = {0.f,0.f,0.f,0.f,0.f,0.f,0.f,0.f};
  acc[0]=zero; acc[1]=zero; acc[2]=zero; acc[3]=zero;

  // chunk t:  k = t/8 (constant per chunk), channel base = (t%8)*32
  auto fill = [&](int t, int buf) {
    const int k     = t >> 3;
    const int cBase = (t & 7) << 5;
    // ---- A: deformable gather; params loaded ONCE per chunk (k fixed) ----
    {
      const int*   id = sIdx[arow*KK_ + k];
      const float* wq = sWt [arow*KK_ + k];
      const int i0 = id[0], i1 = id[1], i2 = id[2], i3 = id[3];
      const float w0 = wq[0], w1 = wq[1], w2 = wq[2], w3 = wq[3];
      const float* xp = x + sXBase[arow] + ((cBase + agrp*8) << 12);
      #pragma unroll
      for (int j = 0; j < 8; ++j) {
        const float* p = xp + (j << 12);               // next channel plane
        As[buf][arow][agrp*8 + j] =
            f2bf(w0*p[i0] + w1*p[i1] + w2*p[i2] + w3*p[i3]);
      }
    }
    // ---- B: pre-packed bf16, two 16B vector loads ----
    {
      const unsigned short* wp =
          wpk + (coBase + brow)*CK_ + (k << 8) + cBase + bhalf*16;
      const uint4* v = (const uint4*)wp;
      uint4* d = (uint4*)&Bs[buf][brow][bhalf*16];
      d[0] = v[0];
      d[1] = v[1];
      __builtin_prefetch(wp + BK, 0, 1);               // next chunk's tile
    }
  };

  fill(0, 0);
  __syncthreads();

  for (int t = 0; t < NCHUNK; ++t) {
    const int cur = t & 1;
    if (t + 1 < NCHUNK)
      fill(t + 1, cur ^ 1);                            // overlap with WMMA below

    // ---- WMMA: 4 x v_wmma_f32_16x16x32_bf16 per wave per chunk ----
    v16bf bfrag;
    {
      int coL = (wv << 4) + ln16;                      // B lane: N=lane%16, K=hs*8+e and 16+hs*8+e
      #pragma unroll
      for (int e = 0; e < 8; ++e) { BFU u; u.u = Bs[cur][coL][hs*8 + e]; bfrag[e] = u.b; u.u = Bs[cur][coL][16 + hs*8 + e]; bfrag[8 + e] = u.b; }
    }
    #pragma unroll
    for (int mt = 0; mt < 4; ++mt) {
      v16bf afrag;                                     // A lane: M = lane%16
      int row = (mt << 4) + ln16;
      #pragma unroll
      for (int e = 0; e < 8; ++e) {                    // K = hs*8+e and 16+hs*8+e
        BFU u;
        u.u = As[cur][row][hs*8 + e];       afrag[e]     = u.b;
        u.u = As[cur][row][16 + hs*8 + e];  afrag[8 + e] = u.b;
      }
      acc[mt] = __builtin_amdgcn_wmma_f32_16x16x32_bf16(
          false, afrag, false, bfrag, (short)0, acc[mt], false, false);
    }
    __syncthreads();   // single barrier per chunk (double-buffered)
  }

  // ---- epilogue: scatter f32 accumulators to NCHW out ----
  {
    int co = coBase + (wv << 4) + ln16;                // D lane: N = lane%16
    #pragma unroll
    for (int mt = 0; mt < 4; ++mt) {
      #pragma unroll
      for (int r = 0; r < 8; ++r) {
        int pl = (mt << 4) + (hs << 3) + r;            // M = r + 8*(lane/16)
        out[sOutBase[pl] + (co << 12)] = acc[mt][r];
      }
    }
  }
}

extern "C" void kernel_launch(void* const* d_in, const int* in_sizes, int n_in,
                              void* d_out, int out_size, void* d_ws, size_t ws_size,
                              hipStream_t stream) {
  const float* x      = (const float*)d_in[0];
  const float* offset = (const float*)d_in[1];
  const float* mask   = (const float*)d_in[2];
  const float* weight = (const float*)d_in[3];
  float* out = (float*)d_out;
  unsigned short* wpk = (unsigned short*)d_ws;   // 256*2304*2 B = 1.18 MB

  // 1) repack weight to bf16, k-major reduction order (deterministic each call)
  int total = CO_ * CK_;
  pack_weight_bf16<<<(total + 255) / 256, 256, 0, stream>>>(weight, wpk);

  // 2) fused deformable-sample + WMMA GEMM
  dim3 grid((N_*HO_*WO_)/BM, CO_/BN, 1);   // 512 x 2
  dim3 block(256, 1, 1);                   // 8 wave32 waves
  dcn_fused_wmma<<<grid, block, 0, stream>>>(x, offset, mask, wpk, out);

  (void)in_sizes; (void)n_in; (void)out_size; (void)ws_size;
}
